// GAT_33998961115431
// MI455X (gfx1250) — compile-verified, matrix operand order checked by reading, not run on hardware
//
#include <hip/hip_runtime.h>
#include <hip/hip_bf16.h>
#include <math.h>

typedef __attribute__((ext_vector_type(16))) __bf16 v16bf;
typedef __attribute__((ext_vector_type(8)))  float  v8f;

#define GEMM_K 128   // F_in / hidden width is 128 for all three layers

union Frag { v16bf v; uint4 q[2]; };

__device__ __forceinline__ unsigned short f2bf(float f) {
  // round-to-nearest-even fp32 -> bf16
  unsigned u = __float_as_uint(f);
  unsigned r = u + 0x7FFFu + ((u >> 16) & 1u);
  return (unsigned short)(r >> 16);
}

__device__ __forceinline__ float leaky(float x) { return x > 0.f ? x : 0.2f * x; }

// float atomic max via sign-split integer atomics (init value must be -inf)
__device__ __forceinline__ void atomicMaxF(float* addr, float v) {
  if (v >= 0.f) atomicMax((int*)addr, __float_as_int(v));
  else          atomicMin((unsigned int*)addr, __float_as_uint(v));
}

#if defined(__has_builtin)
#if __has_builtin(__builtin_amdgcn_global_load_async_to_lds_b128)
#define HAVE_ASYNC_LDS 1
#endif
#if __has_builtin(__builtin_amdgcn_s_wait_asynccnt)
#define HAVE_WAIT_ASYNC 1
#endif
#endif

// fp32 -> bf16 elementwise convert
__global__ void gat_cvt_bf16(const float* __restrict__ src,
                             unsigned short* __restrict__ dst, long n) {
  long i = blockIdx.x * (long)blockDim.x + threadIdx.x;
  if (i < n) dst[i] = f2bf(src[i]);
}

// fp32 W[K x Ncol] -> bf16 W^T[Ncol x K] (column-major W, rows contiguous in K)
__global__ void gat_cvt_wT(const float* __restrict__ W,
                           unsigned short* __restrict__ wT, int K, int Ncol) {
  int i = blockIdx.x * blockDim.x + threadIdx.x;
  if (i >= K * Ncol) return;
  int k = i / Ncol, c = i % Ncol;
  wT[(size_t)c * K + k] = f2bf(W[i]);
}

// ---------------------------------------------------------------------------
// WMMA GEMM: Hout[M x Ncol] = Xbf[M x 128] @ (WT[Ncol x 128])^T, f32 accum.
// One block = one 16-row stripe; wave w = column tile w. A stripe staged in
// LDS (4KB, async-to-LDS when available), fragments read as b128 loads.
// 16-bit A lane layout: lane L holds row M=L%16; VGPRs 0-3 = K k0+half*8..+7,
// VGPRs 4-7 = K k0+16+half*8..+7 (half = L/16) -> two contiguous 16B runs.
// B (32x16) mirrors with N = L%16, so W^T rows give the same two 16B runs.
// ---------------------------------------------------------------------------
__global__ void __launch_bounds__(256) gat_wmma_gemm(
    const unsigned short* __restrict__ Xbf,
    const unsigned short* __restrict__ WT,
    float* __restrict__ Hout, int Ncol, int tilesN) {
  __shared__ __align__(16) unsigned short sA[16 * GEMM_K];  // 4KB stripe
  int t = threadIdx.x;
  int tm = blockIdx.x;

  // stage A stripe: 16 rows x 128 bf16 = 4096B = 256 threads x 16B
  const uint4* gsrc = (const uint4*)(Xbf + (size_t)tm * 16 * GEMM_K);
#if defined(HAVE_ASYNC_LDS)
  typedef __attribute__((ext_vector_type(4))) int i4v;
  typedef __attribute__((address_space(1))) i4v* as1v4p;
  typedef __attribute__((address_space(3))) i4v* as3v4p;
  __builtin_amdgcn_global_load_async_to_lds_b128(
      (as1v4p)(gsrc + t), (as3v4p)((uint4*)sA + t), 0, 0);
#if defined(HAVE_WAIT_ASYNC)
  __builtin_amdgcn_s_wait_asynccnt(0);
#endif
#else
  ((uint4*)sA)[t] = gsrc[t];
#endif
  __syncthreads();

  int wv = t >> 5;
  if (wv >= tilesN) return;              // wave-uniform
  int lane = t & 31;
  int half = lane >> 4, l16 = lane & 15;
  int col = wv * 16 + l16;               // B column for this lane

  const unsigned short* arow = sA + l16 * GEMM_K;          // LDS, row M=l16
  const unsigned short* brow = WT + (size_t)col * GEMM_K;  // global, col N
  v8f acc = {};
#pragma unroll
  for (int k0 = 0; k0 < GEMM_K; k0 += 32) {
    Frag a, b;
    a.q[0] = *(const uint4*)(arow + k0 + half * 8);        // ds_load_b128
    a.q[1] = *(const uint4*)(arow + k0 + 16 + half * 8);
    b.q[0] = *(const uint4*)(brow + k0 + half * 8);        // global_load_b128
    b.q[1] = *(const uint4*)(brow + k0 + 16 + half * 8);
    acc = __builtin_amdgcn_wmma_f32_16x16x32_bf16(
        false, a.v, false, b.v, (short)0, acc, false, false);
  }
#pragma unroll
  for (int r = 0; r < 8; ++r) {
    int orow = tm * 16 + r + half * 8;   // C/D layout: VGPR r -> M = r + half*8
    Hout[(size_t)orow * Ncol + col] = acc[r];
  }
}

// alpha_s[n,h] = dot(h[n,h,:], a_src[h,:]);  alpha_d likewise
__global__ void gat_alpha(const float* __restrict__ Hf,
                          const float* __restrict__ Asrc,
                          const float* __restrict__ Adst,
                          float* __restrict__ as_, float* __restrict__ ad_,
                          int Nn, int Hh, int C) {
  int idx = blockIdx.x * blockDim.x + threadIdx.x;
  if (idx >= Nn * Hh) return;
  int n = idx / Hh, h = idx % Hh;
  const float* hp = Hf + (size_t)n * Hh * C + (size_t)h * C;
  float ss = 0.f, sd = 0.f;
  for (int c = 0; c < C; ++c) {
    float v = hp[c];
    ss += v * Asrc[h * C + c];
    sd += v * Adst[h * C + c];
  }
  as_[idx] = ss;
  ad_[idx] = sd;
}

__global__ void gat_fill(float* __restrict__ p, float v, long n) {
  long i = blockIdx.x * (long)blockDim.x + threadIdx.x;
  if (i < n) p[i] = v;
}

// edge pass 1: segment max over dst of leaky(as[src]+ad[dst])
__global__ void gat_edge_max(const int* __restrict__ srcp, const int* __restrict__ dstp,
                             const float* __restrict__ as_, const float* __restrict__ ad_,
                             float* __restrict__ m_, int E, int Nn, int Hh) {
  long total = (long)(E + Nn) * Hh;
  for (long i = blockIdx.x * (long)blockDim.x + threadIdx.x; i < total;
       i += (long)gridDim.x * blockDim.x) {
    int e = (int)(i / Hh), h = (int)(i % Hh);
    int s = e < E ? srcp[e] : e - E;
    int d = e < E ? dstp[e] : e - E;
    float v = leaky(as_[s * Hh + h] + ad_[d * Hh + h]);
    atomicMaxF(&m_[d * Hh + h], v);
  }
}

// edge pass 2: denom = segment_sum(exp(e - m[dst]))
__global__ void gat_edge_sum(const int* __restrict__ srcp, const int* __restrict__ dstp,
                             const float* __restrict__ as_, const float* __restrict__ ad_,
                             const float* __restrict__ m_, float* __restrict__ dn_,
                             int E, int Nn, int Hh) {
  long total = (long)(E + Nn) * Hh;
  for (long i = blockIdx.x * (long)blockDim.x + threadIdx.x; i < total;
       i += (long)gridDim.x * blockDim.x) {
    int e = (int)(i / Hh), h = (int)(i % Hh);
    int s = e < E ? srcp[e] : e - E;
    int d = e < E ? dstp[e] : e - E;
    float v = leaky(as_[s * Hh + h] + ad_[d * Hh + h]);
    unsafeAtomicAdd(&dn_[d * Hh + h], __expf(v - m_[d * Hh + h]));
  }
}

// edge pass 3: out[dst] += alpha * h[src]; one wave32 per edge, 32 ch / step
__global__ void __launch_bounds__(256) gat_edge_aggr(
    const int* __restrict__ srcp, const int* __restrict__ dstp,
    const float* __restrict__ as_, const float* __restrict__ ad_,
    const float* __restrict__ m_, const float* __restrict__ dn_,
    const float* __restrict__ Hf, float* __restrict__ out,
    int E, int Nn, int Hh, int logC, int HC) {
  int ET = E + Nn;
  int lane = threadIdx.x & 31;
  int nWaves = gridDim.x * (blockDim.x >> 5);
  int wave = blockIdx.x * (blockDim.x >> 5) + (threadIdx.x >> 5);
  for (int e = wave; e < ET; e += nWaves) {
    int s = e < E ? srcp[e] : e - E;
    int d = e < E ? dstp[e] : e - E;
    const float* hs = Hf + (size_t)s * HC;
    float* od = out + (size_t)d * HC;
    __builtin_prefetch(hs + lane, 0, 0);   // global_prefetch_b8 of h[src] row
    int lastHead = -1;
    float alpha = 0.f;
    for (int j = lane; j < HC; j += 32) {
      int h = j >> logC;
      if (h != lastHead) {
        float v = leaky(as_[s * Hh + h] + ad_[d * Hh + h]);
        alpha = __expf(v - m_[d * Hh + h]) / (dn_[d * Hh + h] + 1e-16f);
        lastHead = h;
      }
      unsafeAtomicAdd(&od[j], alpha * hs[j]);   // hw global_atomic_add_f32
    }
  }
}

__global__ void gat_bias(float* __restrict__ out, const float* __restrict__ b,
                         long total, int HC) {
  long i = blockIdx.x * (long)blockDim.x + threadIdx.x;
  if (i < total) out[i] += b[i % HC];
}

// reference returns (x, x): duplicate final features into d_out
__global__ void gat_final(const float* __restrict__ src, float* __restrict__ dst, int n) {
  int i = blockIdx.x * blockDim.x + threadIdx.x;
  if (i < n) { dst[i] = src[i]; dst[n + i] = src[i]; }
}

extern "C" void kernel_launch(void* const* d_in, const int* in_sizes, int n_in,
                              void* d_out, int out_size, void* d_ws, size_t ws_size,
                              hipStream_t stream) {
  const float* x   = (const float*)d_in[0];
  const int*   ei  = (const int*)d_in[1];
  const float* W0  = (const float*)d_in[2];
  const float* A0s = (const float*)d_in[3];
  const float* A0d = (const float*)d_in[4];
  const float* b0  = (const float*)d_in[5];
  const float* W1  = (const float*)d_in[6];
  const float* A1s = (const float*)d_in[7];
  const float* A1d = (const float*)d_in[8];
  const float* b1  = (const float*)d_in[9];
  const float* W2  = (const float*)d_in[10];
  const float* A2s = (const float*)d_in[11];
  const float* A2d = (const float*)d_in[12];
  const float* b2  = (const float*)d_in[13];

  const int FIN = 128;
  int N = in_sizes[0] / FIN;      // 50000 (multiple of 16)
  int E = in_sizes[1] / 2;        // 800000
  const int* srcp = ei;
  const int* dstp = ei + E;

  float* ws   = (float*)d_ws;
  float* hbuf = ws;                               // N*128 f32
  float* xo   = hbuf + (size_t)N * 128;           // N*128 f32 (out -> next in)
  float* as_  = xo + (size_t)N * 128;             // N*8
  float* ad_  = as_ + (size_t)N * 8;              // N*8
  float* m_   = ad_ + (size_t)N * 8;              // N*8
  float* dn_  = m_ + (size_t)N * 8;               // N*8
  unsigned short* xbf = (unsigned short*)(dn_ + (size_t)N * 8);  // N*128 bf16
  unsigned short* wT  = xbf + (size_t)N * 128;                   // 128*128 bf16

  auto runLayer = [&](const float* xin, const float* W, const float* aS,
                      const float* aD, const float* bias, int Hh, int C) {
    int HC = Hh * C;
    int tilesN = HC / 16;
    long xn = (long)N * GEMM_K;
    gat_cvt_bf16<<<(int)((xn + 255) / 256), 256, 0, stream>>>(xin, xbf, xn);
    int wn = GEMM_K * HC;
    gat_cvt_wT<<<(wn + 255) / 256, 256, 0, stream>>>(W, wT, GEMM_K, HC);
    gat_wmma_gemm<<<N / 16, 256, 0, stream>>>(xbf, wT, hbuf, HC, tilesN);

    int nh = N * Hh;
    gat_alpha<<<(nh + 255) / 256, 256, 0, stream>>>(hbuf, aS, aD, as_, ad_, N, Hh, C);
    gat_fill<<<(nh + 255) / 256, 256, 0, stream>>>(m_, -INFINITY, nh);
    gat_fill<<<(nh + 255) / 256, 256, 0, stream>>>(dn_, 0.f, nh);
    long outN = (long)N * HC;
    gat_fill<<<(int)((outN + 255) / 256), 256, 0, stream>>>(xo, 0.f, outN);

    int blocks = 8192;
    gat_edge_max<<<blocks, 256, 0, stream>>>(srcp, dstp, as_, ad_, m_, E, N, Hh);
    gat_edge_sum<<<blocks, 256, 0, stream>>>(srcp, dstp, as_, ad_, m_, dn_, E, N, Hh);
    int logC = (C == 16) ? 4 : ((C == 64) ? 6 : 7);
    gat_edge_aggr<<<4096, 256, 0, stream>>>(srcp, dstp, as_, ad_, m_, dn_, hbuf,
                                            xo, E, N, Hh, logC, HC);
    gat_bias<<<(int)((outN + 255) / 256), 256, 0, stream>>>(xo, bias, outN, HC);
  };

  runLayer(x,  W0, A0s, A0d, b0, 8, 16);    // 128 -> 8x16, concat -> 128
  runLayer(xo, W1, A1s, A1d, b1, 1, 128);   // 128 -> 128
  runLayer(xo, W2, A2s, A2d, b2, 1, 64);    // 128 -> 64, mean over 1 head

  int n = N * 64;
  gat_final<<<(n + 255) / 256, 256, 0, stream>>>(xo, (float*)d_out, n);
}
